// mLSTM_22591527977561
// MI455X (gfx1250) — compile-verified, matrix operand order checked by reading, not run on hardware
//
#include <hip/hip_runtime.h>
#include <math.h>

#define LSEQ 512
#define NB   32
#define HID  1024
#define H4   4096

typedef __bf16 bf16_t;
typedef bf16_t v8bf  __attribute__((ext_vector_type(8)));
typedef bf16_t v16bf __attribute__((ext_vector_type(16)));
typedef float  v8f   __attribute__((ext_vector_type(8)));

static __device__ __forceinline__ v16bf cat8(v8bf lo, v8bf hi) {
    return __builtin_shufflevector(lo, hi, 0,1,2,3,4,5,6,7,8,9,10,11,12,13,14,15);
}

static __device__ __forceinline__ float sigmoidf(float x) {
    return 1.0f / (1.0f + __expf(-x));
}

// One wave computes a 32x16 f32 tile:  out[r,c] = sum_k A[r,k] * W[colbase+c, k]
// bf16 WMMA per-lane K pattern (ISA 7.12.2): element e of lane L covers
//   K = kb + (L>>4)*8 + (e&7) + (e>>3)*16   -> two contiguous 8-element (16B) chunks.
static __device__ __forceinline__ void wmma_tile_32x16(
    const bf16_t* __restrict__ A, int lda,
    const bf16_t* __restrict__ W, int ldw, int colbase,
    int K, v8f& acc0, v8f& acc1)
{
    const int lane = (int)(threadIdx.x & 31u);   // wave32 on gfx1250
    const int half = lane >> 4;
    const int l15  = lane & 15;
    const bf16_t* a0 = A + (size_t)l15        * (size_t)lda;
    const bf16_t* a1 = A + (size_t)(16 + l15) * (size_t)lda;
    const bf16_t* b  = W + (size_t)(colbase + l15) * (size_t)ldw;
    for (int kb = 0; kb < K; kb += 32) {
        const int o = kb + half * 8;
        __builtin_prefetch(b + kb + 256, 0, 0);   // global_prefetch_b8
        v16bf fa0 = cat8(*(const v8bf*)(a0 + o), *(const v8bf*)(a0 + o + 16));
        v16bf fa1 = cat8(*(const v8bf*)(a1 + o), *(const v8bf*)(a1 + o + 16));
        v16bf fb  = cat8(*(const v8bf*)(b  + o), *(const v8bf*)(b  + o + 16));
        acc0 = __builtin_amdgcn_wmma_f32_16x16x32_bf16(false, fa0, false, fb,
                                                       (short)0, acc0, false, false);
        acc1 = __builtin_amdgcn_wmma_f32_16x16x32_bf16(false, fa1, false, fb,
                                                       (short)0, acc1, false, false);
    }
}

// ---------- precompute / utility kernels ----------

__global__ void f32_to_bf16_kernel(const float* __restrict__ src,
                                   bf16_t* __restrict__ dst, int n) {
    for (int i = (int)(blockIdx.x * blockDim.x + threadIdx.x); i < n;
         i += (int)(gridDim.x * blockDim.x))
        dst[i] = (bf16_t)src[i];
}

__global__ void zero_state_kernel(bf16_t* __restrict__ hz,
                                  float* __restrict__ c, int n) {
    for (int i = (int)(blockIdx.x * blockDim.x + threadIdx.x); i < n;
         i += (int)(gridDim.x * blockDim.x)) {
        hz[i] = (bf16_t)0.0f;
        c[i]  = 0.0f;
    }
}

__global__ void copy_f32_kernel(const float* __restrict__ src,
                                float* __restrict__ dst, int n) {
    for (int i = (int)(blockIdx.x * blockDim.x + threadIdx.x); i < n;
         i += (int)(gridDim.x * blockDim.x))
        dst[i] = src[i];
}

// Big parallel input-projection GEMM, register-blocked 32x64 per wave:
//   Y[r, c] = sum_k X[r,k]*W[c,k] + bias[c]
// grid = (Mrows/32, Ncols/512), block = 256 (8 waves x 64 cols).
// A fragments are reused across 4 column tiles -> 8 WMMA per 12 B128 loads.
__global__ __launch_bounds__(256)
void xproj_gemm_kernel(const bf16_t* __restrict__ X,   // Mrows x K
                       const bf16_t* __restrict__ W,   // Ncols x K
                       const float*  __restrict__ bias,
                       float* __restrict__ Y,          // Mrows x Ncols
                       int Ncols, int K)
{
    const int mblk    = (int)blockIdx.x;
    const int wave    = (int)(threadIdx.x >> 5);
    const int colbase = (int)blockIdx.y * 512 + wave * 64;
    const int lane = (int)(threadIdx.x & 31u);
    const int half = lane >> 4, l15 = lane & 15;

    const bf16_t* a0 = X + ((size_t)mblk * 32 + (size_t)l15) * (size_t)K;
    const bf16_t* a1 = a0 + (size_t)16 * (size_t)K;
    const bf16_t* bp[4];
#pragma unroll
    for (int j = 0; j < 4; ++j)
        bp[j] = W + (size_t)(colbase + j * 16 + l15) * (size_t)K;

    v8f acc0[4] = {{0}, {0}, {0}, {0}};
    v8f acc1[4] = {{0}, {0}, {0}, {0}};

    for (int kb = 0; kb < K; kb += 32) {
        const int o = kb + half * 8;
        v16bf fa0 = cat8(*(const v8bf*)(a0 + o), *(const v8bf*)(a0 + o + 16));
        v16bf fa1 = cat8(*(const v8bf*)(a1 + o), *(const v8bf*)(a1 + o + 16));
#pragma unroll
        for (int j = 0; j < 4; ++j) {
            v16bf fb = cat8(*(const v8bf*)(bp[j] + o), *(const v8bf*)(bp[j] + o + 16));
            acc0[j] = __builtin_amdgcn_wmma_f32_16x16x32_bf16(false, fa0, false, fb,
                                                              (short)0, acc0[j], false, false);
            acc1[j] = __builtin_amdgcn_wmma_f32_16x16x32_bf16(false, fa1, false, fb,
                                                              (short)0, acc1[j], false, false);
        }
    }

#pragma unroll
    for (int j = 0; j < 4; ++j) {
        const int col  = colbase + j * 16 + l15;
        const float bv = bias[col];
        float* y0 = Y + ((size_t)mblk * 32 + (size_t)(half * 8)) * (size_t)Ncols + col;
#pragma unroll
        for (int v = 0; v < 8; ++v) {
            y0[(size_t)v * Ncols]        = acc0[j][v] + bv;
            y0[(size_t)(v + 16) * Ncols] = acc1[j][v] + bv;
        }
    }
}

// ---------- per-timestep recurrent kernels ----------
// Kept at one 32x16 tile per wave: the 512 serial steps are L2-latency bound
// (weights L2-resident), so many independent waves in flight > per-wave blocking.

// m = Xm_t * (h_prev @ Wmh^T + bmh);  grid = HID/64, block = 128 (4 waves)
__global__ __launch_bounds__(128)
void mstep_kernel(const bf16_t* __restrict__ hprev,  // NB x HID
                  const bf16_t* __restrict__ Wmh,    // HID x HID
                  const float*  __restrict__ bmh,
                  const float*  __restrict__ Xm_t,   // NB x HID (bias folded)
                  bf16_t* __restrict__ m_out)        // NB x HID
{
    const int wave    = (int)(threadIdx.x >> 5);
    const int colbase = (int)blockIdx.x * 64 + wave * 16;
    v8f acc0 = {}, acc1 = {};
    wmma_tile_32x16(hprev, HID, Wmh, HID, colbase, HID, acc0, acc1);
    const int lane = (int)(threadIdx.x & 31u);
    const int half = lane >> 4, l15 = lane & 15;
    const int col  = colbase + l15;
    const float bv = bmh[col];
#pragma unroll
    for (int v = 0; v < 8; ++v) {
        const int r0 = half * 8 + v, r1 = r0 + 16;
        const float m0 = Xm_t[r0 * HID + col] * (acc0[v] + bv);
        const float m1 = Xm_t[r1 * HID + col] * (acc1[v] + bv);
        m_out[r0 * HID + col] = (bf16_t)m0;
        m_out[r1 * HID + col] = (bf16_t)m1;
    }
}

// g = Xg_t + m @ Wgm^T (bias folded into Xg); gate math fused via LDS.
// grid = HID/16, block = 128 (wave w handles gate w for the same 16 hidden cols)
__global__ __launch_bounds__(128)
void gatestep_kernel(const bf16_t* __restrict__ m_in,  // NB x HID
                     const bf16_t* __restrict__ Wgm,   // H4 x HID
                     const float*  __restrict__ Xg_t,  // NB x H4
                     float*  __restrict__ c_state,     // NB x HID (in/out)
                     float*  __restrict__ y_f32,       // NB x HID (h_t, f32)
                     bf16_t* __restrict__ h_bf_next)   // NB x HID (h_t, bf16)
{
    __shared__ float g_s[4][NB][16];
    const int gate    = (int)(threadIdx.x >> 5);
    const int colbase = (int)blockIdx.x * 16;
    v8f acc0 = {}, acc1 = {};
    wmma_tile_32x16(m_in, HID, Wgm + (size_t)gate * HID * HID, HID, colbase, HID,
                    acc0, acc1);
    const int lane = (int)(threadIdx.x & 31u);
    const int half = lane >> 4, l15 = lane & 15;
    const int gcol = gate * HID + colbase + l15;
#pragma unroll
    for (int v = 0; v < 8; ++v) {
        const int r0 = half * 8 + v, r1 = r0 + 16;
        g_s[gate][r0][l15] = acc0[v] + Xg_t[r0 * H4 + gcol];
        g_s[gate][r1][l15] = acc1[v] + Xg_t[r1 * H4 + gcol];
    }
    __syncthreads();
    for (int idx = (int)threadIdx.x; idx < NB * 16; idx += (int)blockDim.x) {
        const int n = idx >> 4, c = idx & 15;
        const int col = colbase + c;
        const float ig = g_s[0][n][c], fg = g_s[1][n][c];
        const float zg = g_s[2][n][c], og = g_s[3][n][c];
        const float cn = sigmoidf(fg) * c_state[n * HID + col]
                       + sigmoidf(ig) * tanhf(zg);
        const float hn = sigmoidf(og) * tanhf(cn);
        c_state[n * HID + col]   = cn;
        y_f32[n * HID + col]     = hn;
        h_bf_next[n * HID + col] = (bf16_t)hn;
    }
}

// ---------- host orchestration ----------

extern "C" void kernel_launch(void* const* d_in, const int* in_sizes, int n_in,
                              void* d_out, int out_size, void* d_ws, size_t ws_size,
                              hipStream_t stream)
{
    (void)in_sizes; (void)n_in; (void)out_size; (void)ws_size;

    const float* inputs  = (const float*)d_in[0];
    const float* Wmx[2]  = {(const float*)d_in[1],  (const float*)d_in[8]};
    const float* Wmh[2]  = {(const float*)d_in[2],  (const float*)d_in[9]};
    const float* bmx[2]  = {(const float*)d_in[3],  (const float*)d_in[10]};
    const float* bmh[2]  = {(const float*)d_in[4],  (const float*)d_in[11]};
    const float* Wgx[2]  = {(const float*)d_in[5],  (const float*)d_in[12]};
    const float* Wgm[2]  = {(const float*)d_in[6],  (const float*)d_in[13]};
    const float* bg [2]  = {(const float*)d_in[7],  (const float*)d_in[14]};

    float* out    = (float*)d_out;
    float* ys_out = out;                                        // (L,N,H)
    float* hs_out = out + (size_t)LSEQ * NB * HID;              // (2,N,H)
    float* cs_out = hs_out + (size_t)2 * NB * HID;              // (2,N,H)

    char* ws = (char*)d_ws;
    size_t cur = 0;
    auto alloc = [&](size_t bytes) -> void* {
        void* p = ws + cur;
        cur += (bytes + 255) & ~(size_t)255;
        return p;
    };

    const size_t NH  = (size_t)NB * HID;       // 32768
    const size_t LNH = (size_t)LSEQ * NH;      // 16.7M

    bf16_t* x0_bf  = (bf16_t*)alloc(LNH * 2);
    bf16_t* y0_bf  = (bf16_t*)alloc(LNH * 2);
    bf16_t* Wmx_bf[2], *Wmh_bf[2], *Wgx_bf[2], *Wgm_bf[2];
    for (int l = 0; l < 2; ++l) {
        Wmx_bf[l] = (bf16_t*)alloc((size_t)HID * HID * 2);
        Wmh_bf[l] = (bf16_t*)alloc((size_t)HID * HID * 2);
        Wgx_bf[l] = (bf16_t*)alloc((size_t)H4  * HID * 2);
        Wgm_bf[l] = (bf16_t*)alloc((size_t)H4  * HID * 2);
    }
    float*  Xm      = (float*) alloc(LNH * 4);                       // (L,N,H)
    float*  Xg      = (float*) alloc((size_t)LSEQ * NB * H4 * 4);    // (L,N,4H)
    bf16_t* hz_bf   = (bf16_t*)alloc(NH * 2);
    bf16_t* m_bf    = (bf16_t*)alloc(NH * 2);
    bf16_t* h1_bf   = (bf16_t*)alloc(NH * 2);
    float*  c_state = (float*) alloc(NH * 4);
    float*  h_f32   = (float*) alloc(NH * 4);

    // 1) convert inputs + weights to bf16
    f32_to_bf16_kernel<<<2048, 256, 0, stream>>>(inputs, x0_bf, (int)LNH);
    for (int l = 0; l < 2; ++l) {
        f32_to_bf16_kernel<<<1024, 256, 0, stream>>>(Wmx[l], Wmx_bf[l], HID * HID);
        f32_to_bf16_kernel<<<1024, 256, 0, stream>>>(Wmh[l], Wmh_bf[l], HID * HID);
        f32_to_bf16_kernel<<<2048, 256, 0, stream>>>(Wgx[l], Wgx_bf[l], H4 * HID);
        f32_to_bf16_kernel<<<2048, 256, 0, stream>>>(Wgm[l], Wgm_bf[l], H4 * HID);
    }
    zero_state_kernel<<<64, 256, 0, stream>>>(hz_bf, c_state, (int)NH);

    const int Mblocks = (LSEQ * NB) / 32;   // 512

    // ================= layer 0 =================
    xproj_gemm_kernel<<<dim3(Mblocks, HID / 512), 256, 0, stream>>>(
        x0_bf, Wmx_bf[0], bmx[0], Xm, HID, HID);
    xproj_gemm_kernel<<<dim3(Mblocks, H4 / 512), 256, 0, stream>>>(
        x0_bf, Wgx_bf[0], bg[0], Xg, H4, HID);

    for (int t = 0; t < LSEQ; ++t) {
        const bf16_t* hprev = (t == 0) ? hz_bf : (y0_bf + (size_t)(t - 1) * NH);
        mstep_kernel<<<HID / 64, 128, 0, stream>>>(
            hprev, Wmh_bf[0], bmh[0], Xm + (size_t)t * NH, m_bf);
        gatestep_kernel<<<HID / 16, 128, 0, stream>>>(
            m_bf, Wgm_bf[0], Xg + (size_t)t * NB * H4,
            c_state, h_f32, y0_bf + (size_t)t * NH);
    }
    copy_f32_kernel<<<64, 256, 0, stream>>>(h_f32,   hs_out, (int)NH);
    copy_f32_kernel<<<64, 256, 0, stream>>>(c_state, cs_out, (int)NH);

    // ================= layer 1 (state carried over, reference quirk) =========
    xproj_gemm_kernel<<<dim3(Mblocks, HID / 512), 256, 0, stream>>>(
        y0_bf, Wmx_bf[1], bmx[1], Xm, HID, HID);
    xproj_gemm_kernel<<<dim3(Mblocks, H4 / 512), 256, 0, stream>>>(
        y0_bf, Wgx_bf[1], bg[1], Xg, H4, HID);

    for (int t = 0; t < LSEQ; ++t) {
        const bf16_t* hprev = (t == 0) ? (y0_bf + (size_t)(LSEQ - 1) * NH) : h1_bf;
        mstep_kernel<<<HID / 64, 128, 0, stream>>>(
            hprev, Wmh_bf[1], bmh[1], Xm + (size_t)t * NH, m_bf);
        gatestep_kernel<<<HID / 16, 128, 0, stream>>>(
            m_bf, Wgm_bf[1], Xg + (size_t)t * NB * H4,
            c_state, ys_out + (size_t)t * NH, h1_bf);
    }
    copy_f32_kernel<<<64, 256, 0, stream>>>(ys_out + (size_t)(LSEQ - 1) * NH,
                                            hs_out + NH, (int)NH);
    copy_f32_kernel<<<64, 256, 0, stream>>>(c_state, cs_out + NH, (int)NH);
}
